// EdgeDecoder_24515673325798
// MI455X (gfx1250) — matrix operand order repaired and checked by reading, not compiled, not run on hardware
//
#include <hip/hip_runtime.h>
#include <hip/hip_bf16.h>

typedef __attribute__((ext_vector_type(16))) __bf16 v16bf;
typedef __attribute__((ext_vector_type(8)))  float  v8f;

#define HIDDEN 128
#define KDIM   256            // 2*HIDDEN (concat)
#define TILE_M 16
#define WAVES_PER_BLOCK 8
#define THREADS (WAVES_PER_BLOCK * 32)

// W1 (fp32 [128 out][256 in]) is staged into LDS as bf16, pre-swizzled into
// WMMA B-fragment order: 64 fragments (ks=0..7 x nt=0..7), each 1024 bytes,
// lane l's 8 dwords (16 bf16 = one v16bf B operand) at byte offset l*32.
// B[k][n] = W1[n][k] ; fragment (ks,nt): n = nt*16 + (l&15),
// k = ks*32 + (l>>4)*16 + 2j (+1) for dword j=0..7.
__global__ __launch_bounds__(THREADS)
void edge_decoder_kernel(const float* __restrict__ z_user,
                         const float* __restrict__ z_movie,
                         const long long* __restrict__ edge_index,
                         const float* __restrict__ W1,
                         const float* __restrict__ b1,
                         const float* __restrict__ W2,
                         const float* __restrict__ b2,
                         float* __restrict__ out,
                         int n_edge)
{
    __shared__ __align__(32) unsigned lds_w1[64 * 256];   // 64 KB

    const int tid  = threadIdx.x;
    const int lane = tid & 31;
    const int wave = tid >> 5;

    // ---- Stage W1 -> LDS (fp32 -> packed bf16, B-fragment swizzle) ----
    #pragma unroll 4
    for (int i = 0; i < 64; ++i) {
        int idx  = tid + i * THREADS;          // 0..16383 dwords
        int frag = idx >> 8;                   // 0..63
        int wi   = idx & 255;
        int l    = wi >> 3;                    // fragment lane
        int j    = wi & 7;                     // dword within lane
        int ks   = frag >> 3;
        int nt   = frag & 7;
        int n    = nt * 16 + (l & 15);
        int k0   = ks * 32 + ((l >> 4) << 4) + (j << 1);
        float f0 = W1[n * KDIM + k0];
        float f1 = W1[n * KDIM + k0 + 1];
        union { __bf16 h[2]; unsigned u; } pk;
        pk.h[0] = (__bf16)f0;
        pk.h[1] = (__bf16)f1;
        lds_w1[idx] = pk.u;
    }

    // Per-lane second-layer params for column n = lane&15 of each n-tile.
    const int ncol = lane & 15;
    float b1v[8], w2v[8];
    #pragma unroll
    for (int nt = 0; nt < 8; ++nt) {
        b1v[nt] = b1[nt * 16 + ncol];
        w2v[nt] = W2[nt * 16 + ncol];
    }
    const float bias2 = b2[0];

    __syncthreads();

    const int n_tiles = n_edge / TILE_M;
    const int gwave   = blockIdx.x * WAVES_PER_BLOCK + wave;
    const int n_gwave = gridDim.x * WAVES_PER_BLOCK;

    const int m     = lane & 15;          // edge row within tile
    const int khalf = (lane >> 4) << 4;   // 0 or 16 (A-matrix k-half)
    const unsigned lane_b_off = (unsigned)(lane * 32);

    for (int tile = gwave; tile < n_tiles; tile += n_gwave) {
        const int e = tile * TILE_M + m;
        const long long rowi = edge_index[e];
        const long long coli = edge_index[n_edge + e];
        const float* __restrict__ zu = z_user  + rowi * HIDDEN;
        const float* __restrict__ zm = z_movie + coli * HIDDEN;

        v8f acc[8];
        #pragma unroll
        for (int nt = 0; nt < 8; ++nt)
            #pragma unroll
            for (int r = 0; r < 8; ++r) acc[nt][r] = 0.0f;

        #pragma unroll
        for (int ks = 0; ks < 8; ++ks) {
            const int kbase = ks * 32;
            const float* src = (kbase < HIDDEN) ? (zu + kbase)
                                                : (zm + (kbase - HIDDEN));
            // A fragment: lane row m, K = kbase+khalf+0..15
            const float4 fa = *(const float4*)(src + khalf);
            const float4 fb = *(const float4*)(src + khalf + 4);
            const float4 fc = *(const float4*)(src + khalf + 8);
            const float4 fd = *(const float4*)(src + khalf + 12);
            v16bf a;
            a[0]  = (__bf16)fa.x; a[1]  = (__bf16)fa.y;
            a[2]  = (__bf16)fa.z; a[3]  = (__bf16)fa.w;
            a[4]  = (__bf16)fb.x; a[5]  = (__bf16)fb.y;
            a[6]  = (__bf16)fb.z; a[7]  = (__bf16)fb.w;
            a[8]  = (__bf16)fc.x; a[9]  = (__bf16)fc.y;
            a[10] = (__bf16)fc.z; a[11] = (__bf16)fc.w;
            a[12] = (__bf16)fd.x; a[13] = (__bf16)fd.y;
            a[14] = (__bf16)fd.z; a[15] = (__bf16)fd.w;

            // One opaque base offset per ks-step: stops LICM from hoisting
            // the 64 fragment loads out of the tile loop (which previously
            // spilled them to scratch), while letting the 8 nt-fragments be
            // addressed via DS immediate offsets off a single VGPR.
            unsigned boff = (unsigned)(ks * 8 * 1024) + lane_b_off;
            asm volatile("" : "+v"(boff));
            const char* bbase = (const char*)lds_w1 + boff;

            #pragma unroll
            for (int nt = 0; nt < 8; ++nt) {
                const v16bf bfrag = *(const v16bf*)(bbase + nt * 1024);
                acc[nt] = __builtin_amdgcn_wmma_f32_16x16x32_bf16(
                    false, a, false, bfrag, (short)0, acc[nt], false, false);
            }
        }

        // ---- Epilogue: h = relu(acc + b1); logit = h . w2 + b2; 5*sigmoid ----
        // C/D layout: VGPR r, lane l -> M = r + 8*(l>=16), N = l&15.
        float res[8];
        #pragma unroll
        for (int r = 0; r < 8; ++r) {
            float s = 0.0f;
            #pragma unroll
            for (int nt = 0; nt < 8; ++nt) {
                float h = acc[nt][r] + b1v[nt];
                h = h > 0.0f ? h : 0.0f;
                s += h * w2v[nt];
            }
            // reduce across the 16 lanes sharing this M-row
            s += __shfl_xor(s, 1, 32);
            s += __shfl_xor(s, 2, 32);
            s += __shfl_xor(s, 4, 32);
            s += __shfl_xor(s, 8, 32);
            res[r] = 5.0f / (1.0f + __expf(-(s + bias2)));
        }
        if ((lane & 15) == 0) {
            float* o = out + tile * TILE_M + ((lane >> 4) << 3);
            #pragma unroll
            for (int r = 0; r < 8; ++r) o[r] = res[r];
        }
    }
}

extern "C" void kernel_launch(void* const* d_in, const int* in_sizes, int n_in,
                              void* d_out, int out_size, void* d_ws, size_t ws_size,
                              hipStream_t stream) {
    const float*     z_user     = (const float*)d_in[0];
    const float*     z_movie    = (const float*)d_in[1];
    const long long* edge_index = (const long long*)d_in[2];
    const float*     W1         = (const float*)d_in[3];
    const float*     b1         = (const float*)d_in[4];
    const float*     W2         = (const float*)d_in[5];
    const float*     b2         = (const float*)d_in[6];
    float* out = (float*)d_out;
    const int n_edge = in_sizes[2] / 2;   // edge_index is [2, E] int64

    const int blocks = 1024;
    hipLaunchKernelGGL(edge_decoder_kernel, dim3(blocks), dim3(THREADS), 0, stream,
                       z_user, z_movie, edge_index, W1, b1, W2, b2, out, n_edge);
}
